// GCN_89481348645293
// MI455X (gfx1250) — matrix operand order repaired and checked by reading, not compiled, not run on hardware
//
#include <hip/hip_runtime.h>

// ---------------------------------------------------------------------------
// GCN forward for MI455X (gfx1250, wave32).
// GEMMs run on v_wmma_f32_16x16x32_f16. fp32 operands are converted to f16
// and stored into LDS in *fragment order* (inverse of the ISA 7.12.2 per-lane
// layouts), so each wave reads its A/B fragments with packed ds_load_b128
// instead of per-half scalar gathers.
// ---------------------------------------------------------------------------

typedef __attribute__((ext_vector_type(16))) _Float16 v16h;
typedef __attribute__((ext_vector_type(4)))  _Float16 v4h;
typedef __attribute__((ext_vector_type(8)))  float    v8f;

#define NCLS 10
#define GFEAT 32
#define GSTEP 14

// -------------------------------- utility ----------------------------------
__global__ void zero_f32(float* __restrict__ p, long long n) {
  long long i = (long long)blockIdx.x * blockDim.x + threadIdx.x;
  if (i < n) p[i] = 0.0f;
}

__global__ void deg_count(const int* __restrict__ dst, float* __restrict__ deg, long long E) {
  long long i = (long long)blockIdx.x * blockDim.x + threadIdx.x;
  if (i < E) atomicAdd(&deg[dst[i]], 1.0f);
}

__global__ void deg_to_dis(float* __restrict__ deg, int N) {
  int i = blockIdx.x * blockDim.x + threadIdx.x;
  if (i < N) deg[i] = rsqrtf(deg[i] + 1.0f);
}

// ----------------------- WMMA GEMM: Y[M,N] = X[M,K] @ W[K,N] ---------------
// Block = 256 threads = 8 waves. Block tile 64(M) x 32(N), K-step 32.
// Wave w: mi = w&3 -> 16-row sub-tile, ni = w>>2 -> 16-col sub-tile.
//
// LDS holds tiles in fragment order:
//   A slot: ((mi*32 + 16*lhi + m) * 16 + h), where for K-index k within the
//           32-wide step: lhi = (k>>3)&1, h = (k&7) + 8*(k>>4)
//   B slot: ((ni*32 + 16*lhi + n) * 16 + h), with lhi and h as for A
// so each lane's v16h fragment is 32 contiguous bytes.
__global__ __launch_bounds__(256) void gemm_f16_wmma(
    const float* __restrict__ X, const float* __restrict__ W,
    float* __restrict__ Y, int M, int K, int N)
{
  __shared__ __align__(32) _Float16 As[64 * 32];   // 4 KB, fragment order
  __shared__ __align__(32) _Float16 Bs[32 * 32];   // 2 KB, fragment order

  const int tid  = threadIdx.x;
  const int lane = tid & 31;
  const int wave = tid >> 5;
  const int lhi  = lane >> 4;
  const int lm   = lane & 15;

  const int mi = wave & 3;
  const int ni = wave >> 2;

  const int m0 = blockIdx.x * 64;
  const int n0 = blockIdx.y * 32;

  v8f acc = {};

  for (int kk = 0; kk < K; kk += 32) {
    // ---- stage A: 64x32 fp32 -> f16, 512 float4 groups, 2 per thread ----
#pragma unroll
    for (int g = tid; g < 512; g += 256) {
      const int r = g >> 3;              // 0..63
      const int c = (g & 7) << 2;        // 0,4,...,28 (4 consecutive K)
      const float4 q = *(const float4*)(X + (long long)(m0 + r) * K + kk + c);
      const int lh = (c >> 3) & 1;
      const int hb = (c & 7) + ((c >> 4) << 3);
      const int idx = (((r >> 4) << 5) + (lh << 4) + (r & 15)) * 16 + hb;
      v4h p = { (_Float16)q.x, (_Float16)q.y, (_Float16)q.z, (_Float16)q.w };
      *(v4h*)(As + idx) = p;             // ds_store_b64
    }
    // ---- stage B: 32x32 fp32 -> f16, 256 groups of 4 K-rows, 1/thread ----
    {
      const int cn = tid & 31;           // column within tile
      const int kr = (tid >> 5) << 2;    // 0,4,...,28
      const float* wp = W + (long long)(kk + kr) * N + n0 + cn;
      v4h p = { (_Float16)wp[0], (_Float16)wp[N],
                (_Float16)wp[2 * N], (_Float16)wp[3 * N] };
      const int idx = (((cn >> 4) << 5) + (((kr >> 3) & 1) << 4) + (cn & 15)) * 16 + (kr & 7) + ((kr >> 4) << 3);
      *(v4h*)(Bs + idx) = p;             // ds_store_b64
    }
    __syncthreads();

    // packed fragment reads: 32 contiguous bytes per lane (2x ds_load_b128)
    v16h a = *(const v16h*)(As + ((mi << 5) + lane) * 16);
    v16h b = *(const v16h*)(Bs + ((ni << 5) + lane) * 16);

    // (neg_a, A, neg_b, B, c_mod, C, reuse_a, reuse_b)
    acc = __builtin_amdgcn_wmma_f32_16x16x32_f16(false, a, false, b,
                                                 (short)0, acc, false, false);
    __syncthreads();
  }

  // C/D layout: VGPR r, lanes 0-15 -> M=r, lanes 16-31 -> M=8+r; N = lane%16
  const int row0 = m0 + mi * 16 + 8 * lhi;
  const int col  = n0 + ni * 16 + lm;
#pragma unroll
  for (int r = 0; r < 8; ++r)
    Y[(long long)(row0 + r) * N + col] = acc[r];
}

// -------------------- GCN aggregation: D^-1/2 (A+I) D^-1/2 h + b -----------
__global__ void self_loop_bias(const float* __restrict__ tmp, const float* __restrict__ dis,
                               const float* __restrict__ bias, float* __restrict__ agg,
                               int Nn, int C) {
  long long idx = (long long)blockIdx.x * blockDim.x + threadIdx.x;
  if (idx >= (long long)Nn * C) return;
  int i = (int)(idx / C), c = (int)(idx % C);
  float d = dis[i];
  agg[idx] = d * d * tmp[idx] + bias[c];
}

__global__ void edge_agg(const float* __restrict__ tmp, const int* __restrict__ src,
                         const int* __restrict__ dst, const float* __restrict__ dis,
                         float* __restrict__ agg, long long E, int C) {
  long long idx = (long long)blockIdx.x * blockDim.x + threadIdx.x;
  if (idx >= E * (long long)C) return;
  long long e = idx / C;
  int c = (int)(idx % C);
  int s = src[e], d = dst[e];
  float w = dis[s] * dis[d];
  atomicAdd(&agg[(long long)d * C + c], w * tmp[(long long)s * C + c]);
}

// ------------------------------ batchnorm ----------------------------------
__global__ __launch_bounds__(256) void bn_stats(const float* __restrict__ x,
                                                const float* __restrict__ g,
                                                const float* __restrict__ bb,
                                                float* __restrict__ scale,
                                                float* __restrict__ shift,
                                                int Nr, int C, int relu_in) {
  const int c = blockIdx.x;
  float s = 0.0f, sq = 0.0f;
  for (int i = threadIdx.x; i < Nr; i += 256) {
    float v = x[(long long)i * C + c];
    if (relu_in) v = fmaxf(v, 0.0f);
    s += v; sq += v * v;
  }
  __shared__ float shs[256], shq[256];
  shs[threadIdx.x] = s; shq[threadIdx.x] = sq;
  __syncthreads();
  for (int off = 128; off > 0; off >>= 1) {
    if (threadIdx.x < off) {
      shs[threadIdx.x] += shs[threadIdx.x + off];
      shq[threadIdx.x] += shq[threadIdx.x + off];
    }
    __syncthreads();
  }
  if (threadIdx.x == 0) {
    float mean = shs[0] / (float)Nr;
    float var  = shq[0] / (float)Nr - mean * mean;
    float sc   = g[c] * rsqrtf(var + 1e-5f);
    scale[c] = sc;
    shift[c] = bb[c] - mean * sc;
  }
}

__global__ void bn_apply(float* __restrict__ x, const float* __restrict__ scale,
                         const float* __restrict__ shift, long long total, int C,
                         int relu_in, int relu_out) {
  long long idx = (long long)blockIdx.x * blockDim.x + threadIdx.x;
  if (idx >= total) return;
  int c = (int)(idx % C);
  float v = x[idx];
  if (relu_in) v = fmaxf(v, 0.0f);
  v = v * scale[c] + shift[c];
  if (relu_out) v = fmaxf(v, 0.0f);
  x[idx] = v;
}

// ------------------------------ gate / pool --------------------------------
__global__ void linear_small(const float* __restrict__ x, const float* __restrict__ W,
                             const float* __restrict__ b, float* __restrict__ y,
                             int Nr, int Cin, int Cout) {
  long long idx = (long long)blockIdx.x * blockDim.x + threadIdx.x;
  if (idx >= (long long)Nr * Cout) return;
  int i = (int)(idx / Cout), j = (int)(idx % Cout);
  float acc = b[j];
  for (int k = 0; k < Cin; ++k) acc += x[(long long)i * Cin + k] * W[k * Cout + j];
  y[idx] = acc;
}

__global__ void pool_softmax(const float* __restrict__ h, const float* __restrict__ gate,
                             float* __restrict__ pooled, int B) {
  int idx = blockIdx.x * blockDim.x + threadIdx.x;
  if (idx >= B * GFEAT) return;
  int b = idx / GFEAT, f = idx % GFEAT;
  int base = b * GSTEP;
  float m = -3.402823466e+38f;
  for (int k = 0; k < GSTEP; ++k) m = fmaxf(m, gate[base + k]);
  float ssum = 0.0f, acc = 0.0f;
  for (int k = 0; k < GSTEP; ++k) {
    float e = __expf(gate[base + k] - m);
    ssum += e;
    acc  += e * h[(long long)(base + k) * GFEAT + f];
  }
  pooled[idx] = acc / ssum;
}

__global__ void fc_softmax(const float* __restrict__ pooled, const float* __restrict__ W,
                           const float* __restrict__ b, float* __restrict__ out, int B) {
  int g = blockIdx.x * blockDim.x + threadIdx.x;
  if (g >= B) return;
  float logits[NCLS];
  float m = -3.402823466e+38f;
  for (int j = 0; j < NCLS; ++j) {
    float a = b[j];
    for (int k = 0; k < GFEAT; ++k) a += pooled[(long long)g * GFEAT + k] * W[k * NCLS + j];
    logits[j] = a;
    m = fmaxf(m, a);
  }
  float s = 0.0f;
  for (int j = 0; j < NCLS; ++j) { logits[j] = __expf(logits[j] - m); s += logits[j]; }
  for (int j = 0; j < NCLS; ++j) out[(long long)g * NCLS + j] = logits[j] / s;
}

// ------------------------------- driver ------------------------------------
static inline long long cdiv(long long a, long long b) { return (a + b - 1) / b; }

extern "C" void kernel_launch(void* const* d_in, const int* in_sizes, int n_in,
                              void* d_out, int out_size, void* d_ws, size_t ws_size,
                              hipStream_t stream) {
  (void)n_in; (void)out_size; (void)ws_size;

  const int F_IN = 512;
  const int N    = in_sizes[0] / F_IN;        // 114688
  const long long E = in_sizes[1] / 2;        // 917504
  const int B    = N / GSTEP;                 // 8192

  const float* x   = (const float*)d_in[0];
  const int*   ei  = (const int*)d_in[1];
  const int*   src = ei;
  const int*   dst = ei + E;

  const float* W[4]   = { (const float*)d_in[2],  (const float*)d_in[6],
                          (const float*)d_in[10], (const float*)d_in[14] };
  const float* bia[4] = { (const float*)d_in[3],  (const float*)d_in[7],
                          (const float*)d_in[11], (const float*)d_in[15] };
  const float* bng[4] = { (const float*)d_in[4],  (const float*)d_in[8],
                          (const float*)d_in[12], (const float*)d_in[16] };
  const float* bnb[4] = { (const float*)d_in[5],  (const float*)d_in[9],
                          (const float*)d_in[13], (const float*)d_in[17] };
  const float* gW1   = (const float*)d_in[18];
  const float* gb1   = (const float*)d_in[19];
  const float* gbn1g = (const float*)d_in[20];
  const float* gbn1b = (const float*)d_in[21];
  const float* gW2   = (const float*)d_in[22];
  const float* gb2   = (const float*)d_in[23];
  const float* gbn2g = (const float*)d_in[24];
  const float* gbn2b = (const float*)d_in[25];
  const float* fcW   = (const float*)d_in[26];
  const float* fcb   = (const float*)d_in[27];

  // workspace layout (floats)
  float* f = (float*)d_ws;
  size_t off = 0;
  float* bufA   = f + off; off += (size_t)N * 256;
  float* bufB   = f + off; off += (size_t)N * 256;
  float* dis    = f + off; off += (size_t)N;
  float* gate1  = f + off; off += (size_t)N * 16;
  float* gvec   = f + off; off += (size_t)N;
  float* pooled = f + off; off += (size_t)B * GFEAT;
  float* scale  = f + off; off += 256;
  float* shift  = f + off; off += 256;

  const int T = 256;

  // --- degree -> dis = rsqrt(deg+1) ---
  zero_f32<<<cdiv(N, T), T, 0, stream>>>(dis, N);
  deg_count<<<cdiv(E, T), T, 0, stream>>>(dst, dis, E);
  deg_to_dis<<<cdiv(N, T), T, 0, stream>>>(dis, N);

  // --- four GCN layers ---
  const int dims[5] = { 512, 128, 256, 64, 32 };
  const float* cur = x;
  for (int l = 0; l < 4; ++l) {
    const int Cin = dims[l], Cout = dims[l + 1];
    dim3 grid(N / 64, Cout / 32);
    gemm_f16_wmma<<<grid, T, 0, stream>>>(cur, W[l], bufA, N, Cin, Cout);
    self_loop_bias<<<cdiv((long long)N * Cout, T), T, 0, stream>>>(bufA, dis, bia[l], bufB, N, Cout);
    edge_agg<<<cdiv(E * (long long)Cout, T), T, 0, stream>>>(bufA, src, dst, dis, bufB, E, Cout);
    bn_stats<<<Cout, T, 0, stream>>>(bufB, bng[l], bnb[l], scale, shift, N, Cout, /*relu_in=*/1);
    bn_apply<<<cdiv((long long)N * Cout, T), T, 0, stream>>>(bufB, scale, shift,
                                                             (long long)N * Cout, Cout, 1, 0);
    cur = bufB;
  }

  // --- gate network: Linear -> BN -> ReLU -> Linear -> BN -> ReLU ---
  linear_small<<<cdiv((long long)N * 16, T), T, 0, stream>>>(bufB, gW1, gb1, gate1, N, 32, 16);
  bn_stats<<<16, T, 0, stream>>>(gate1, gbn1g, gbn1b, scale, shift, N, 16, /*relu_in=*/0);
  bn_apply<<<cdiv((long long)N * 16, T), T, 0, stream>>>(gate1, scale, shift,
                                                         (long long)N * 16, 16, 0, 1);
  linear_small<<<cdiv((long long)N, T), T, 0, stream>>>(gate1, gW2, gb2, gvec, N, 16, 1);
  bn_stats<<<1, T, 0, stream>>>(gvec, gbn2g, gbn2b, scale, shift, N, 1, /*relu_in=*/0);
  bn_apply<<<cdiv((long long)N, T), T, 0, stream>>>(gvec, scale, shift, (long long)N, 1, 0, 1);

  // --- segment softmax pooling + FC + softmax ---
  pool_softmax<<<cdiv((long long)B * GFEAT, T), T, 0, stream>>>(bufB, gvec, pooled, B);
  fc_softmax<<<cdiv(B, T), T, 0, stream>>>(pooled, fcW, fcb, (float*)d_out, B);
}
